// VectorQuantizer_25503515804103
// MI455X (gfx1250) — matrix operand order repaired and checked by reading, not compiled, not run on hardware
//
#include <hip/hip_runtime.h>

typedef __attribute__((ext_vector_type(16))) _Float16 v16h;
typedef __attribute__((ext_vector_type(8)))  _Float16 v8h;
typedef __attribute__((ext_vector_type(8)))  float    v8f;

#define N_TOK 65536
#define K_CB  1024
#define D_DIM 256
#define EPS_F 1e-8f

// ---------------------------------------------------------------------------
// Kernel 1/2: L2-normalize each row of a [rows, 256] f32 matrix, emit f16.
// ---------------------------------------------------------------------------
__global__ __launch_bounds__(256) void row_normalize_f16(
    const float* __restrict__ src, _Float16* __restrict__ dst) {
  const int row  = blockIdx.x;
  const int t    = threadIdx.x;
  const float v  = src[(size_t)row * D_DIM + t];
  float ss = v * v;
  #pragma unroll
  for (int o = 16; o > 0; o >>= 1) ss += __shfl_xor(ss, o, 32);
  __shared__ float wsum[8];
  if ((t & 31) == 0) wsum[t >> 5] = ss;
  __syncthreads();
  float tot = 0.f;
  #pragma unroll
  for (int i = 0; i < 8; ++i) tot += wsum[i];
  const float scale = 1.0f / fmaxf(sqrtf(tot), EPS_F);
  dst[(size_t)row * D_DIM + t] = (_Float16)(v * scale);
}

// ---------------------------------------------------------------------------
// Kernel 3: sims = Xn @ Wn^T via v_wmma_f32_16x16x32_f16, running argmax.
// One wave per 32 rows (two 16-row strips, both A strips in registers);
// 8 waves/block = 256 rows/block. Codebook staged through a double-buffered
// 2x32KB LDS tile shared by all waves. Each B fragment read from LDS feeds
// TWO WMMAs (one per strip) -> 16 B/lane of LDS per WMMA instead of 32.
//
// Fragment layouts (wave32, 16x16x32 f16):
//   A (16x32): lane m=l&15 holds row m; halves 0..15 = K[kb..kb+15],
//              kb = (l>=16)?16:0 (same k order as B).
//   B (32x16): lane n=l&15 holds col n; halves 0..15 = K[ko..ko+15],
//              ko = (l>=16)?16:0  -> 32 contiguous bytes of a Wn row.
//   C (16x16): vgpr r: lanes 0-15 -> M=r, lanes 16-31 -> M=8+r; N = l&15.
// ---------------------------------------------------------------------------
#define MT_ROWS 64                      // codebook rows per LDS mega-tile
#define N_MT    (K_CB / MT_ROWS)        // 16 mega-tiles

__global__ __launch_bounds__(256) void vq_argmax_wmma(
    const _Float16* __restrict__ Xn, const _Float16* __restrict__ Wn,
    int* __restrict__ out_idx, float* __restrict__ out_idx_f) {
  __shared__ _Float16 sm[2][MT_ROWS * D_DIM];   // 2 x 32KB

  const int tid      = threadIdx.x;
  const int lane     = tid & 31;
  const int wave     = tid >> 5;
  const int row_base = (blockIdx.x * 8 + wave) * 32;   // two strips: +0, +16
  const int m        = lane & 15;
  const int hi       = lane >> 4;   // lane group: 0 or 1

  // ---- Load both A strips (2 x 16 rows x D=256) into registers.
  v16h a0[8], a1[8];
  {
    const _Float16* x0 = Xn + (size_t)(row_base + m) * D_DIM + hi * 16;
    const _Float16* x1 = x0 + 16 * D_DIM;
    #pragma unroll
    for (int c = 0; c < 8; ++c) {
      v8h lo0  = *(const v8h*)(x0 + c * 32);
      v8h hi0  = *(const v8h*)(x0 + c * 32 + 8);
      a0[c] = __builtin_shufflevector(lo0, hi0, 0, 1, 2, 3, 4, 5, 6, 7,
                                                8, 9, 10, 11, 12, 13, 14, 15);
      v8h lo1  = *(const v8h*)(x1 + c * 32);
      v8h hi1  = *(const v8h*)(x1 + c * 32 + 8);
      a1[c] = __builtin_shufflevector(lo1, hi1, 0, 1, 2, 3, 4, 5, 6, 7,
                                                8, 9, 10, 11, 12, 13, 14, 15);
    }
  }

  float best0[8], best1[8];
  int   bidx0[8], bidx1[8];
  #pragma unroll
  for (int r = 0; r < 8; ++r) {
    best0[r] = -3.0e38f; bidx0[r] = 0;
    best1[r] = -3.0e38f; bidx1[r] = 0;
  }

  const int n  = lane & 15;
  const int ko = hi * 16;

  // ---- Prologue: stage mega-tile 0 into LDS buffer 0.
  {
    #pragma unroll
    for (int i = 0; i < 8; ++i) {
      v8h v = *(const v8h*)(Wn + (i * 256 + tid) * 8);
      *(v8h*)(&sm[0][(i * 256 + tid) * 8]) = v;
    }
  }
  __syncthreads();

  for (int t = 0; t < N_MT; ++t) {
    const int buf = t & 1;

    // Issue global loads for the next mega-tile early (hidden under 64 WMMAs).
    v8h stage[8];
    if (t < N_MT - 1) {
      const _Float16* wsrc = Wn + (size_t)(t + 1) * MT_ROWS * D_DIM;
      #pragma unroll
      for (int i = 0; i < 8; ++i)
        stage[i] = *(const v8h*)(wsrc + (i * 256 + tid) * 8);
    }

    // Compute: 4 column tiles; each B fragment feeds both A strips.
    #pragma unroll
    for (int q = 0; q < 4; ++q) {
      const _Float16* wb = &sm[buf][(q * 16 + n) * D_DIM + ko];
      v8f acc0 = {};
      v8f acc1 = {};
      #pragma unroll
      for (int c = 0; c < 8; ++c) {
        v16h b = *(const v16h*)(wb + c * 32);
        acc0 = __builtin_amdgcn_wmma_f32_16x16x32_f16(
            false, a0[c], false, b, (short)0, acc0, false, false);
        acc1 = __builtin_amdgcn_wmma_f32_16x16x32_f16(
            false, a1[c], false, b, (short)0, acc1, false, false);
      }
      const int ci = t * MT_ROWS + q * 16 + n;
      #pragma unroll
      for (int r = 0; r < 8; ++r) {
        if (acc0[r] > best0[r]) { best0[r] = acc0[r]; bidx0[r] = ci; }
      }
      #pragma unroll
      for (int r = 0; r < 8; ++r) {
        if (acc1[r] > best1[r]) { best1[r] = acc1[r]; bidx1[r] = ci; }
      }
    }

    // Stage next tile into the other buffer. The barrier at the end of the
    // previous iteration guarantees all waves finished reading it.
    if (t < N_MT - 1) {
      #pragma unroll
      for (int i = 0; i < 8; ++i)
        *(v8h*)(&sm[buf ^ 1][(i * 256 + tid) * 8]) = stage[i];
      __syncthreads();
    }
  }

  // ---- Cross-lane argmax within each 16-lane group (first-index tie-break).
  #pragma unroll
  for (int r = 0; r < 8; ++r) {
    float bv = best0[r];
    int   bi = bidx0[r];
    #pragma unroll
    for (int o = 1; o < 16; o <<= 1) {
      float ov = __shfl_xor(bv, o, 32);
      int   oi = __shfl_xor(bi, o, 32);
      if (ov > bv || (ov == bv && oi < bi)) { bv = ov; bi = oi; }
    }
    if (n == 0) {
      const int row = row_base + hi * 8 + r;  // lanes 0/16: rows 0-7 / 8-15
      out_idx[row]   = bi;
      out_idx_f[row] = (float)bi;
    }
  }
  #pragma unroll
  for (int r = 0; r < 8; ++r) {
    float bv = best1[r];
    int   bi = bidx1[r];
    #pragma unroll
    for (int o = 1; o < 16; o <<= 1) {
      float ov = __shfl_xor(bv, o, 32);
      int   oi = __shfl_xor(bi, o, 32);
      if (ov > bv || (ov == bv && oi < bi)) { bv = ov; bi = oi; }
    }
    if (n == 0) {
      const int row = row_base + 16 + hi * 8 + r;
      out_idx[row]   = bi;
      out_idx_f[row] = (float)bi;
    }
  }
}

// ---------------------------------------------------------------------------
// Kernel 4: gather codebook rows -> quantized output, per-row SSE partials.
// ---------------------------------------------------------------------------
__global__ __launch_bounds__(256) void vq_gather_loss(
    const float* __restrict__ x, const float* __restrict__ weight,
    const int* __restrict__ idx, float* __restrict__ out_q,
    float* __restrict__ partial) {
  const int row  = blockIdx.x;
  const int t    = threadIdx.x;
  const int code = idx[row];
  const float w  = weight[(size_t)code * D_DIM + t];
  const float xv = x[(size_t)row * D_DIM + t];
  out_q[(size_t)row * D_DIM + t] = w;  // straight-through output == gathered code
  const float d = w - xv;
  float ss = d * d;
  #pragma unroll
  for (int o = 16; o > 0; o >>= 1) ss += __shfl_xor(ss, o, 32);
  __shared__ float wsum[8];
  if ((t & 31) == 0) wsum[t >> 5] = ss;
  __syncthreads();
  if (t == 0) {
    float tot = 0.f;
    #pragma unroll
    for (int i = 0; i < 8; ++i) tot += wsum[i];
    partial[row] = tot;
  }
}

// ---------------------------------------------------------------------------
// Kernel 5: deterministic final loss reduction (single block, no atomics).
// loss = 0.25*q_latent + 0.25*e_latent = 0.5 * mean((q-x)^2)
// ---------------------------------------------------------------------------
__global__ __launch_bounds__(256) void vq_loss_reduce(
    const float* __restrict__ partial, float* __restrict__ loss) {
  float s = 0.f;
  for (int i = threadIdx.x; i < N_TOK; i += 256) s += partial[i];
  #pragma unroll
  for (int o = 16; o > 0; o >>= 1) s += __shfl_xor(s, o, 32);
  __shared__ float wsum[8];
  if ((threadIdx.x & 31) == 0) wsum[threadIdx.x >> 5] = s;
  __syncthreads();
  if (threadIdx.x == 0) {
    float tot = 0.f;
    #pragma unroll
    for (int i = 0; i < 8; ++i) tot += wsum[i];
    *loss = 0.5f * tot / ((float)N_TOK * (float)D_DIM);
  }
}

// ---------------------------------------------------------------------------
extern "C" void kernel_launch(void* const* d_in, const int* in_sizes, int n_in,
                              void* d_out, int out_size, void* d_ws, size_t ws_size,
                              hipStream_t stream) {
  const float* x = (const float*)d_in[0];   // [64,1024,256] f32
  const float* w = (const float*)d_in[1];   // [1024,256]   f32

  // Workspace layout
  char* ws = (char*)d_ws;
  _Float16* Wn      = (_Float16*)ws;                                   // 512 KB
  _Float16* Xn      = (_Float16*)(ws + (size_t)K_CB * D_DIM * 2);      // 32 MB
  int*      idx     = (int*)(ws + (size_t)K_CB * D_DIM * 2
                                + (size_t)N_TOK * D_DIM * 2);          // 256 KB
  float*    partial = (float*)((char*)idx + (size_t)N_TOK * 4);        // 256 KB

  // Output layout: quantized [N*D] | loss [1] | indices [N] (as float)
  float* out_q     = (float*)d_out;
  float* loss      = out_q + (size_t)N_TOK * D_DIM;
  float* out_idx_f = loss + 1;

  row_normalize_f16<<<K_CB, 256, 0, stream>>>(w, Wn);
  row_normalize_f16<<<N_TOK, 256, 0, stream>>>(x, Xn);
  vq_argmax_wmma<<<N_TOK / 32 / 8, 256, 0, stream>>>(Xn, Wn, idx, out_idx_f);
  vq_gather_loss<<<N_TOK, 256, 0, stream>>>(x, w, idx, out_q, partial);
  vq_loss_reduce<<<1, 256, 0, stream>>>(partial, loss);
}
